// DCRNNBaseline_83064667504928
// MI455X (gfx1250) — matrix operand order repaired and checked by reading, not compiled, not run on hardware
//
#include <hip/hip_runtime.h>
#include <hip/hip_bf16.h>
#include <stdint.h>

// Problem constants (from reference): B=32, L=12, N=1024, F=2, H=128, HOR=12
#define Bc   32
#define Lc   12
#define Nc   1024
#define Fc   2
#define Hc   128
#define HORc 12
#define G3   (3 * Hc)   // 384 gate columns
#define KC   512        // K-chunk of A staged in LDS (32 x 512 bf16 = 32 KB)

#define AS1 __attribute__((address_space(1)))
#define AS3 __attribute__((address_space(3)))

#if __has_builtin(__builtin_amdgcn_global_load_async_to_lds_b128)
#define HAVE_ASYNC_LDS 1
#endif

typedef __attribute__((ext_vector_type(16))) __bf16 v16bf;
typedef __attribute__((ext_vector_type(8)))  float  v8f;
typedef int v4i_gcc __attribute__((vector_size(16)));   // matches builtin param type

union BF16Frag { unsigned int u[8]; v16bf v; };

__device__ __forceinline__ unsigned short f2bf(float f) {
  unsigned int u = __float_as_uint(f);
  u += 0x7FFFu + ((u >> 16) & 1u);   // round-to-nearest-even
  return (unsigned short)(u >> 16);
}

// A-operand fragment (16x32 bf16 tile, row-major with leading dim `ld` ushorts).
// rowbase points at element (m, 0) of this lane's matrix row.
// ISA 7.12.2 layout: lanes 0-15 hold K={0..7,16..23}, lanes 16-31 K={8..15,24..31}.
__device__ __forceinline__ v16bf load_a_frag(const unsigned short* rowbase, int kh) {
  BF16Frag f;
  const unsigned int* p = (const unsigned int*)rowbase;
#pragma unroll
  for (int i = 0; i < 8; ++i) {
    int idx = (i < 4 ? i : i + 4) + 4 * kh;  // uint index = bf16-pair index
    f.u[i] = p[idx];
  }
  return f.v;
}

// B-operand fragment from K-contiguous ("transposed") storage: colbase points at
// element (k0, n) where consecutive ushorts advance K. Lanes 0-15: K={0..7,16..23},
// lanes 16-31: K={8..15,24..31} (same order as the A fragment).
__device__ __forceinline__ v16bf load_b_frag_T(const unsigned short* colbase, int kh) {
  BF16Frag f;
  const unsigned int* p = (const unsigned int*)colbase;
#pragma unroll
  for (int i = 0; i < 8; ++i) f.u[i] = p[(i < 4 ? i : i + 4) + 4 * kh];
  return f.v;
}

__device__ __forceinline__ float sigmoidf_(float t) {
  return 1.0f / (1.0f + __expf(-t));
}
__device__ __forceinline__ float tanh_fast(float t) {
  float e = __expf(2.0f * t);
  return 1.0f - 2.0f / (e + 1.0f);
}

#define WMMA_BF16(A_, B_, C_) \
  __builtin_amdgcn_wmma_f32_16x16x32_bf16(false, (A_), false, (B_), (short)0, (C_), false, false)

// ---------------------------------------------------------------------------
// Pack / init kernels
// ---------------------------------------------------------------------------
__global__ void pack_A_kernel(const float* __restrict__ A, unsigned short* __restrict__ Abf) {
  int i = blockIdx.x * 256 + threadIdx.x;      // N*N threads
  Abf[i] = f2bf(A[i]);
}

__global__ void pack_WhT_kernel(const float* __restrict__ Wh, unsigned short* __restrict__ WhT) {
  int i = blockIdx.x * 256 + threadIdx.x;      // G3*H threads; WhT[col*H + k]
  int k = i & (Hc - 1);
  int col = i >> 7;
  WhT[i] = f2bf(Wh[(size_t)k * G3 + col]);
}

__global__ void init_h_kernel(float* __restrict__ hf0, unsigned short* __restrict__ hT0) {
  int i = blockIdx.x * 256 + threadIdx.x;      // B*N*H threads
  hf0[i] = 0.0f;
  hT0[i] = 0;
}

// ---------------------------------------------------------------------------
// Fused GraphGRU step:  h_new = GRU( x_t @ W_x,  (A @ h) @ W_h,  h )
// Grid: (N/32 row-blocks, B batches), 256 threads = 8 waves.
// Wave w owns h-column tile j = [16w,16w+16) and gate tiles {w, w+8, w+16}.
// ---------------------------------------------------------------------------
__global__ __launch_bounds__(256)
void dcrnn_step_kernel(const unsigned short* __restrict__ Abf,   // [N,N] bf16 row-major
                       const unsigned short* __restrict__ WhT,   // [3H,H] bf16 (K-contig)
                       const unsigned short* __restrict__ hTc,   // [B,H,N] bf16 (K-contig)
                       unsigned short*       __restrict__ hTn,
                       const float*          __restrict__ hfc,   // [B,N,H] fp32
                       float*                __restrict__ hfn,
                       const float*          __restrict__ x,     // [B,L,N,F]
                       const float*          __restrict__ Wx,    // [F,3H]
                       const float*          __restrict__ bx,    // [3H]
                       const float*          __restrict__ bh,    // [3H]
                       int t)
{
  __shared__ unsigned short ldsA[32 * KC];    // 32 KB: A row-slab chunk
  __shared__ unsigned short ldsAh[32 * Hc];   //  8 KB: Ah tile (bf16)
  __shared__ float          ldsX[32 * Fc];    // x_t rows for this block

  const int tid  = threadIdx.x;
  const int w    = tid >> 5;          // wave 0..7
  const int lane = tid & 31;
  const int nn   = lane & 15;
  const int kh   = lane >> 4;         // K-half select per WMMA layout
  const int r0   = blockIdx.x * 32;   // first node row of this block
  const int b    = blockIdx.y;        // batch
  const int j    = w * 16 + nn;       // h-column owned by this lane

  const v8f z8 = {0.f, 0.f, 0.f, 0.f, 0.f, 0.f, 0.f, 0.f};

  // ---------------- Phase 1: Ah = A[r0:r0+32, :] @ h_b  (K = 1024) -----------
  v8f accA0 = z8, accA1 = z8;
  const size_t colOff = ((size_t)b * Hc + j) * Nc;   // column j of h^T

  for (int kc = 0; kc < Nc; kc += KC) {
    __syncthreads();   // protect LDS slab reuse
#ifdef HAVE_ASYNC_LDS
    // CDNA5 async DMA: global -> LDS without the VGPR round-trip (ASYNCcnt).
#pragma unroll
    for (int it = 0; it < 8; ++it) {
      int idx8 = (it * 256 + tid) * 8;     // ushort index in 32xKC tile
      int row  = idx8 >> 9;                // / KC
      int col  = idx8 & (KC - 1);
      unsigned short* gp =
          const_cast<unsigned short*>(&Abf[(size_t)(r0 + row) * Nc + kc + col]);
      __builtin_amdgcn_global_load_async_to_lds_b128(
          (AS1 v4i_gcc*)gp, (AS3 v4i_gcc*)&ldsA[idx8], 0, 0);
    }
#if __has_builtin(__builtin_amdgcn_s_wait_asynccnt)
    __builtin_amdgcn_s_wait_asynccnt(0);
#else
    asm volatile("s_wait_asynccnt 0" ::: "memory");
#endif
#else
#pragma unroll
    for (int it = 0; it < 8; ++it) {
      int idx8 = (it * 256 + tid) * 8;     // ushort index in 32xKC tile
      int row  = idx8 >> 9;                // / KC
      int col  = idx8 & (KC - 1);
      *(uint4*)&ldsA[idx8] =
          *(const uint4*)&Abf[(size_t)(r0 + row) * Nc + kc + col];
    }
#endif
    if (kc + KC < Nc)   // global_prefetch_b8 of next A chunk
      __builtin_prefetch(&Abf[(size_t)(r0 + (tid >> 3)) * Nc + kc + KC + (tid & 7) * 64], 0, 1);
    __syncthreads();

#pragma unroll 4
    for (int k0 = 0; k0 < KC; k0 += 32) {
      v16bf bfr = load_b_frag_T(&hTc[colOff + kc + k0], kh);
      v16bf a0  = load_a_frag(&ldsA[(0 * 16 + nn) * KC + k0], kh);
      v16bf a1  = load_a_frag(&ldsA[(1 * 16 + nn) * KC + k0], kh);
      accA0 = WMMA_BF16(a0, bfr, accA0);
      accA1 = WMMA_BF16(a1, bfr, accA1);
    }
  }

  // ---------------- Phase 2: h_gates = Ah @ W_h  (K = 128) -------------------
  // C/D layout: VGPR r <-> row (r + 8*kh), lane n <-> col.
#pragma unroll
  for (int r = 0; r < 8; ++r) {
    ldsAh[(0 * 16 + r + 8 * kh) * Hc + j] = f2bf(accA0[r]);
    ldsAh[(1 * 16 + r + 8 * kh) * Hc + j] = f2bf(accA1[r]);
  }
  if (tid < 32 * Fc)
    ldsX[tid] = x[(((size_t)b * Lc + t) * Nc + r0 + (tid >> 1)) * Fc + (tid & 1)];
  __syncthreads();

  v8f aZ0 = z8, aZ1 = z8, aR0 = z8, aR1 = z8, aN0 = z8, aN1 = z8;
#pragma unroll
  for (int kb = 0; kb < Hc; kb += 32) {
    v16bf a0 = load_a_frag(&ldsAh[(0 * 16 + nn) * Hc + kb], kh);
    v16bf a1 = load_a_frag(&ldsAh[(1 * 16 + nn) * Hc + kb], kh);
    v16bf bz = load_b_frag_T(&WhT[(size_t)(16 * (w +  0) + nn) * Hc + kb], kh);
    v16bf br = load_b_frag_T(&WhT[(size_t)(16 * (w +  8) + nn) * Hc + kb], kh);
    v16bf bn = load_b_frag_T(&WhT[(size_t)(16 * (w + 16) + nn) * Hc + kb], kh);
    aZ0 = WMMA_BF16(a0, bz, aZ0);  aZ1 = WMMA_BF16(a1, bz, aZ1);
    aR0 = WMMA_BF16(a0, br, aR0);  aR1 = WMMA_BF16(a1, br, aR1);
    aN0 = WMMA_BF16(a0, bn, aN0);  aN1 = WMMA_BF16(a1, bn, aN1);
  }

  // ---------------- Phase 3: GRU gate math, all in registers -----------------
  const float wx0z = Wx[j],          wx1z = Wx[G3 + j];
  const float wx0r = Wx[Hc + j],     wx1r = Wx[G3 + Hc + j];
  const float wx0n = Wx[2 * Hc + j], wx1n = Wx[G3 + 2 * Hc + j];
  const float bxz = bx[j], bxr = bx[Hc + j], bxn = bx[2 * Hc + j];
  const float bhz = bh[j], bhr = bh[Hc + j], bhn = bh[2 * Hc + j];

#pragma unroll
  for (int rt = 0; rt < 2; ++rt) {
#pragma unroll
    for (int r = 0; r < 8; ++r) {
      int   m    = rt * 16 + r + 8 * kh;
      int   node = r0 + m;
      float zg = rt ? aZ1[r] : aZ0[r];
      float rg = rt ? aR1[r] : aR0[r];
      float ng = rt ? aN1[r] : aN0[r];
      float x0 = ldsX[m * 2], x1 = ldsX[m * 2 + 1];
      float hold = hfc[((size_t)b * Nc + node) * Hc + j];
      float z  = sigmoidf_(x0 * wx0z + x1 * wx1z + bxz + zg + bhz);
      float rr = sigmoidf_(x0 * wx0r + x1 * wx1r + bxr + rg + bhr);
      float nv = tanh_fast(x0 * wx0n + x1 * wx1n + bxn + rr * (ng + bhn));
      float hnew = (1.0f - z) * nv + z * hold;
      hfn[((size_t)b * Nc + node) * Hc + j]  = hnew;
      hTn[((size_t)b * Hc + j) * Nc + node]  = f2bf(hnew);
    }
  }
}

// ---------------------------------------------------------------------------
// Head: y[b,hor,n] = h[b,n,:] @ W_head + b_head
// ---------------------------------------------------------------------------
__global__ __launch_bounds__(256)
void head_kernel(const float* __restrict__ hf, const float* __restrict__ Whead,
                 const float* __restrict__ bhead, float* __restrict__ y)
{
  __shared__ float wl[Hc * HORc];
  __shared__ float bl[HORc];
  int tid = threadIdx.x;
  for (int i = tid; i < Hc * HORc; i += 256) wl[i] = Whead[i];
  if (tid < HORc) bl[tid] = bhead[tid];
  __syncthreads();

  int g = blockIdx.x * 256 + tid;      // b*N + n
  int b = g >> 10, n = g & (Nc - 1);
  float s[HORc];
#pragma unroll
  for (int o = 0; o < HORc; ++o) s[o] = bl[o];
  const float* hp = &hf[(size_t)g * Hc];
#pragma unroll 8
  for (int hh = 0; hh < Hc; ++hh) {
    float v = hp[hh];
#pragma unroll
    for (int o = 0; o < HORc; ++o) s[o] += v * wl[hh * HORc + o];
  }
#pragma unroll
  for (int o = 0; o < HORc; ++o)
    y[((size_t)b * HORc + o) * Nc + n] = s[o];
}

// ---------------------------------------------------------------------------
// Launch
// ---------------------------------------------------------------------------
extern "C" void kernel_launch(void* const* d_in, const int* in_sizes, int n_in,
                              void* d_out, int out_size, void* d_ws, size_t ws_size,
                              hipStream_t stream) {
  (void)in_sizes; (void)n_in; (void)out_size; (void)ws_size;
  const float* x     = (const float*)d_in[0];
  const float* A     = (const float*)d_in[1];
  const float* Wx    = (const float*)d_in[2];
  const float* bx    = (const float*)d_in[3];
  const float* Wh    = (const float*)d_in[4];
  const float* bh    = (const float*)d_in[5];
  const float* Whead = (const float*)d_in[6];
  const float* bhead = (const float*)d_in[7];
  float* y = (float*)d_out;

  char* ws = (char*)d_ws;
  unsigned short* Abf = (unsigned short*)(ws);                     //  2 MB  [N,N] bf16
  unsigned short* WhT = (unsigned short*)(ws + (2ull  << 20));     // 96 KB  [3H,H] bf16
  unsigned short* hT0 = (unsigned short*)(ws + (4ull  << 20));     //  8 MB  [B,H,N] bf16
  unsigned short* hT1 = (unsigned short*)(ws + (12ull << 20));     //  8 MB
  float*          hf0 = (float*)         (ws + (20ull << 20));     // 16 MB  [B,N,H] fp32
  float*          hf1 = (float*)         (ws + (36ull << 20));     // 16 MB

  pack_A_kernel  <<<(Nc * Nc) / 256,      256, 0, stream>>>(A,  Abf);
  pack_WhT_kernel<<<(G3 * Hc) / 256,      256, 0, stream>>>(Wh, WhT);
  init_h_kernel  <<<(Bc * Nc * Hc) / 256, 256, 0, stream>>>(hf0, hT0);

  for (int t = 0; t < Lc; ++t) {
    const unsigned short* hTc = (t & 1) ? hT1 : hT0;
    unsigned short*       hTn = (t & 1) ? hT0 : hT1;
    const float*          hfc = (t & 1) ? hf1 : hf0;
    float*                hfn = (t & 1) ? hf0 : hf1;
    dcrnn_step_kernel<<<dim3(Nc / 32, Bc), 256, 0, stream>>>(
        Abf, WhT, hTc, hTn, hfc, hfn, x, Wx, bx, bh, t);
  }
  // 12 steps -> final h is back in buffer 0
  head_kernel<<<(Bc * Nc) / 256, 256, 0, stream>>>(hf0, Whead, bhead, y);
}
